// attention_layer_9861244912296
// MI455X (gfx1250) — matrix operand order repaired and checked by reading, not compiled, not run on hardware
//
#include <hip/hip_runtime.h>
#include <hip/hip_bf16.h>
#include <math.h>

typedef __bf16 bf16_t;
typedef __attribute__((ext_vector_type(16))) __bf16 v16bf;
typedef __attribute__((ext_vector_type(8)))  __bf16 v8bf;
typedef __attribute__((ext_vector_type(4)))  __bf16 v4bf;
typedef __attribute__((ext_vector_type(8)))  float  v8f;
typedef __attribute__((ext_vector_type(4)))  float  v4f;

union ABFrag { v16bf v; v8bf h[2]; };

#define BB   2
#define SS   2048
#define DD   1024
#define HH   16
#define DHH  64

__device__ __forceinline__ bf16_t to_bf16(float f) { return (bf16_t)f; }

__device__ __forceinline__ v8f wmma_bf16(v16bf a, v16bf b, v8f c) {
  // D = A(16x32 bf16) * B(32x16 bf16) + C(16x16 f32)
  return __builtin_amdgcn_wmma_f32_16x16x32_bf16(
      /*neg_a=*/false, a, /*neg_b=*/false, b,
      /*c_mod=*/(short)0, c, /*reuse_a=*/false, /*reuse_b=*/false);
}

// A fragment: 16x32 tile of a row-major [*, ld] bf16 matrix, top-left (m0,k0).
// Lane l<16: row m0+l, K chunks {0..7,16..23}; lane>=16: row m0+l-16, K {8..15,24..31}.
__device__ __forceinline__ v16bf load_frag_a(const bf16_t* __restrict__ src,
                                             int ld, int m0, int k0, int lane) {
  const int m  = m0 + (lane & 15);
  const int kh = (lane >> 4) * 8;
  const bf16_t* p = src + (size_t)m * (size_t)ld + (size_t)(k0 + kh);
  ABFrag u;
  u.h[0] = *(const v8bf*)(p);
  u.h[1] = *(const v8bf*)(p + 16);
  return u.v;
}

// B fragment: 32x16 (KxN) tile where column n is stored contiguously over k
// (i.e. source is the transposed matrix, row n = column, length ld over k).
// Lane l<16: col n0+l, K chunks {0..7,16..23}; lane>=16: col n0+l-16, K {8..15,24..31}.
__device__ __forceinline__ v16bf load_frag_b(const bf16_t* __restrict__ srcT,
                                             int ld, int n0, int k0, int lane) {
  const int n  = n0 + (lane & 15);
  const int kh = (lane >> 4) * 8;
  const bf16_t* p = srcT + (size_t)n * (size_t)ld + (size_t)(k0 + kh);
  ABFrag u;
  u.h[0] = *(const v8bf*)(p);
  u.h[1] = *(const v8bf*)(p + 16);
  return u.v;
}

// ---------------- vectorized fp32 -> bf16 (float4 in, 4x bf16 out) ----------------
__global__ __launch_bounds__(256) void cvt_bf16_kernel(const float* __restrict__ in,
                                                       bf16_t* __restrict__ out, int n4) {
  int i = blockIdx.x * 256 + threadIdx.x;
  if (i < n4) {
    v4f f = ((const v4f*)in)[i];
    v4bf o;
    o[0] = to_bf16(f[0]); o[1] = to_bf16(f[1]);
    o[2] = to_bf16(f[2]); o[3] = to_bf16(f[3]);
    ((v4bf*)out)[i] = o;
  }
}

// ---------------- LDS-tiled weight transpose + convert: WT[n][k] = W[k][n] ----------
// W: [K=1024][N=1024] f32, WT: [N][K] bf16. 32x32 tiles, 256 threads = 32x8.
// Both the global read and the global write are coalesced.
__global__ __launch_bounds__(256) void wt_transpose_kernel(const float* __restrict__ W,
                                                           bf16_t* __restrict__ WT) {
  __shared__ bf16_t tile[32][33];                 // +1 pad: no bank conflicts
  const int tx = threadIdx.x & 31;
  const int ty = threadIdx.x >> 5;                // 0..7
  const int k0 = blockIdx.x * 32;                 // tile origin in K
  const int n0 = blockIdx.y * 32;                 // tile origin in N
#pragma unroll
  for (int i = 0; i < 4; ++i) {
    const int k = k0 + ty + i * 8;
    tile[ty + i * 8][tx] = to_bf16(W[(size_t)k * DD + (size_t)(n0 + tx)]);
  }
  __syncthreads();
#pragma unroll
  for (int i = 0; i < 4; ++i) {
    const int n = n0 + ty + i * 8;
    WT[(size_t)n * DD + (size_t)(k0 + tx)] = tile[tx][ty + i * 8];
  }
}

// ---------------- QKV projection GEMM ----------------
// Y[m][n] = sum_k X[m][k] * W[k][n] + bias[n],  M=4096, N=1024, K=1024 (bf16 in, f32 acc)
// Output written bf16: layout 0 -> [B,H,S,DH] (for Q,K); layout 1 -> [B,H,DH,S] (for V^T)
__global__ __launch_bounds__(256) void qkv_gemm_kernel(const bf16_t* __restrict__ X,
                                                       const bf16_t* __restrict__ WT,
                                                       const float*  __restrict__ bias,
                                                       bf16_t* __restrict__ out,
                                                       int transposeOut) {
  const int lane = threadIdx.x & 31;
  const int wid  = threadIdx.x >> 5;
  const int wm   = wid >> 1;        // 0..3
  const int wn   = wid & 1;         // 0..1
  const int m0   = blockIdx.x * 128 + wm * 32;
  const int n0   = blockIdx.y * 128 + wn * 64;

  v8f acc[2][4];
#pragma unroll
  for (int i = 0; i < 2; ++i)
#pragma unroll
    for (int j = 0; j < 4; ++j) acc[i][j] = {};

  for (int k0 = 0; k0 < DD; k0 += 32) {
    // prefetch next k-step lines into cache (global_prefetch_b8; no LOADcnt cost)
    if (k0 + 64 < DD) {
      __builtin_prefetch(X  + (size_t)(m0 + (lane & 15)) * DD + (size_t)(k0 + 64), 0, 1);
      __builtin_prefetch(X  + (size_t)(m0 + 16 + (lane & 15)) * DD + (size_t)(k0 + 64), 0, 1);
      __builtin_prefetch(WT + (size_t)(n0 + (lane & 31)) * DD + (size_t)(k0 + 64), 0, 1);
      __builtin_prefetch(WT + (size_t)(n0 + 32 + (lane & 31)) * DD + (size_t)(k0 + 64), 0, 1);
    }
    v16bf a0 = load_frag_a(X, DD, m0,      k0, lane);
    v16bf a1 = load_frag_a(X, DD, m0 + 16, k0, lane);
#pragma unroll
    for (int j = 0; j < 4; ++j) {
      v16bf bfr = load_frag_b(WT, DD, n0 + j * 16, k0, lane);
      acc[0][j] = wmma_bf16(a0, bfr, acc[0][j]);
      acc[1][j] = wmma_bf16(a1, bfr, acc[1][j]);
    }
  }

  const int hi8 = (lane >> 4) * 8;
#pragma unroll
  for (int i = 0; i < 2; ++i) {
#pragma unroll
    for (int j = 0; j < 4; ++j) {
      const int nb = n0 + j * 16 + (lane & 15);     // global output column
      const int h  = nb >> 6;                       // head
      const int dh = nb & 63;                       // dim within head
      const float bi = bias[nb];
#pragma unroll
      for (int r = 0; r < 8; ++r) {
        const int m = m0 + i * 16 + r + hi8;        // global output row
        const int b = m >> 11;                      // batch (S = 2048)
        const int s = m & (SS - 1);
        const bf16_t val = to_bf16(acc[i][j][r] + bi);
        if (!transposeOut) {
          out[(((size_t)(b * HH + h) * SS + s) * DHH) + dh] = val;          // [B,H,S,DH]
        } else {
          out[(((size_t)(b * HH + h) * DHH + dh) * SS) + s] = val;          // [B,H,DH,S]
        }
      }
    }
  }
}

// ---------------- flash attention ----------------
// grid: (S/128, B*H); block: 256 threads = 8 waves; each wave owns 16 query rows.
__global__ __launch_bounds__(256) void flash_attn_kernel(const bf16_t* __restrict__ Qb,  // [B,H,S,DH]
                                                         const bf16_t* __restrict__ Kb,  // [B,H,S,DH]
                                                         const bf16_t* __restrict__ Vt,  // [B,H,DH,S]
                                                         const int*    __restrict__ mask,// [B,S,S]
                                                         float*        __restrict__ out) // [B,S,H*DH]
{
  __shared__ float  s_sc[8][16][64];     // per-wave raw scores (fp32)
  __shared__ bf16_t s_p [8][16][64];     // per-wave probs (bf16, A-frag staging)
  __shared__ float  s_st[8][3][16];      // per-wave row stats: [0]=max [1]=sum [2]=scale

  const int lane = threadIdx.x & 31;
  const int wid  = threadIdx.x >> 5;
  const int bh   = blockIdx.y;           // 0..31
  const int b    = bh >> 4;
  const int h    = bh & 15;
  const int q0   = blockIdx.x * 128 + wid * 16;   // query row within head

  const bf16_t* Qh = Qb + (size_t)bh * SS * DHH;
  const bf16_t* Kh = Kb + (size_t)bh * SS * DHH;
  const bf16_t* Vh = Vt + (size_t)bh * DHH * SS;

  float (*scp)[64] = s_sc[wid];
  bf16_t (*pp)[64] = s_p[wid];
  float* st_max = s_st[wid][0];
  float* st_sum = s_st[wid][1];
  float* st_scl = s_st[wid][2];

  if (lane < 16) { st_max[lane] = -3.0e38f; st_sum[lane] = 0.0f; st_scl[lane] = 1.0f; }
  __syncthreads();

  // Q fragments for the two 32-wide K-steps over DH=64
  v16bf qf0 = load_frag_a(Qh, DHH, q0, 0,  lane);
  v16bf qf1 = load_frag_a(Qh, DHH, q0, 32, lane);

  v8f o[4];
#pragma unroll
  for (int j = 0; j < 4; ++j) o[j] = {};

  const int hi8 = (lane >> 4) * 8;

  for (int kb = 0; kb < SS; kb += 64) {
    // prefetch next key block of K and V^T
    if (kb + 64 < SS) {
      __builtin_prefetch(Kh + (size_t)(kb + 64 + (lane & 31)) * DHH, 0, 1);
      __builtin_prefetch(Kh + (size_t)(kb + 96 + (lane & 31)) * DHH, 0, 1);
      __builtin_prefetch(Vh + (size_t)(lane & 31) * SS + (size_t)(kb + 64), 0, 1);
      __builtin_prefetch(Vh + (size_t)(32 + (lane & 31)) * SS + (size_t)(kb + 64), 0, 1);
    }
    // ---- scores S = Q * K^T (4 tiles of 16 keys, contraction over DH) ----
#pragma unroll
    for (int j = 0; j < 4; ++j) {
      v8f sc = {};
      v16bf kb0 = load_frag_b(Kh, DHH, kb + j * 16, 0,  lane);
      sc = wmma_bf16(qf0, kb0, sc);
      v16bf kb1 = load_frag_b(Kh, DHH, kb + j * 16, 32, lane);
      sc = wmma_bf16(qf1, kb1, sc);
      const int n = j * 16 + (lane & 15);
#pragma unroll
      for (int r = 0; r < 8; ++r)
        scp[r + hi8][n] = sc[r] * 0.125f;          // 1/sqrt(64)
    }
    __syncthreads();

    // ---- masked online softmax over this 64-key block ----
    {
      const int row = lane >> 1;
      const int c0  = (lane & 1) * 32;
      const int* mrow = mask + ((size_t)b * SS + (size_t)(q0 + row)) * SS + kb + c0;
      float lm = -3.0e38f;
#pragma unroll 8
      for (int c = 0; c < 32; ++c) {
        float x = scp[row][c0 + c] + (1.0f - (float)mrow[c]) * -10000.0f;
        scp[row][c0 + c] = x;
        lm = fmaxf(lm, x);
      }
      lm = fmaxf(lm, __shfl_xor(lm, 1, 32));
      const float oldm = st_max[row];
      const float nm   = fmaxf(oldm, lm);
      float psum = 0.0f;
#pragma unroll 8
      for (int c = 0; c < 32; ++c) {
        float p = __expf(scp[row][c0 + c] - nm);
        psum += p;
        pp[row][c0 + c] = to_bf16(p);
      }
      psum += __shfl_xor(psum, 1, 32);
      const float scl = __expf(oldm - nm);
      if ((lane & 1) == 0) {
        st_max[row] = nm;
        st_sum[row] = st_sum[row] * scl + psum;
        st_scl[row] = scl;
      }
    }
    __syncthreads();

    // ---- rescale O, then O += P * V (contraction over the 64 keys) ----
    float sclv[8];
#pragma unroll
    for (int r = 0; r < 8; ++r) sclv[r] = st_scl[r + hi8];
#pragma unroll
    for (int j = 0; j < 4; ++j)
#pragma unroll
      for (int r = 0; r < 8; ++r) o[j][r] *= sclv[r];

#pragma unroll
    for (int k0 = 0; k0 < 64; k0 += 32) {
      // P A-fragment straight from LDS staging
      const int prow = lane & 15;
      const int kh   = (lane >> 4) * 8;
      ABFrag u;
      u.h[0] = *(const v8bf*)&pp[prow][k0 + kh];
      u.h[1] = *(const v8bf*)&pp[prow][k0 + 16 + kh];
      v16bf pf = u.v;
#pragma unroll
      for (int j = 0; j < 4; ++j) {
        v16bf vb = load_frag_b(Vh, SS, j * 16, kb + k0, lane);  // V^T rows = dh, contiguous keys
        o[j] = wmma_bf16(pf, vb, o[j]);
      }
    }
    __syncthreads();
  }

  // ---- normalize and store: out[b][s][h*64+dh] ----
  float rsum[8];
#pragma unroll
  for (int r = 0; r < 8; ++r) rsum[r] = st_sum[r + hi8];
#pragma unroll
  for (int j = 0; j < 4; ++j) {
    const int col = h * 64 + j * 16 + (lane & 15);
#pragma unroll
    for (int r = 0; r < 8; ++r) {
      const int m = q0 + r + hi8;
      out[((size_t)b * SS + m) * (HH * DHH) + col] = o[j][r] / rsum[r];
    }
  }
}

extern "C" void kernel_launch(void* const* d_in, const int* in_sizes, int n_in,
                              void* d_out, int out_size, void* d_ws, size_t ws_size,
                              hipStream_t stream) {
  const float* from_t = (const float*)d_in[0];   // [B,S,D]
  const float* to_t   = (const float*)d_in[1];   // [B,S,D]
  const int*   amask  = (const int*)  d_in[2];   // [B,S,S]
  const float* Wq     = (const float*)d_in[3];
  const float* bq     = (const float*)d_in[4];
  const float* Wk     = (const float*)d_in[5];
  const float* bk     = (const float*)d_in[6];
  const float* Wv     = (const float*)d_in[7];
  const float* bv     = (const float*)d_in[8];
  float* out = (float*)d_out;

  const size_t MK  = (size_t)BB * SS * DD;       // 4096*1024 activations
  const size_t KN  = (size_t)DD * DD;            // 1024*1024 weights
  char* ws = (char*)d_ws;
  bf16_t* Xf  = (bf16_t*)(ws);                               // from, bf16
  bf16_t* Xt  = (bf16_t*)(ws + 2 * MK);                      // to, bf16
  bf16_t* WqT = (bf16_t*)(ws + 4 * MK);                      // [N,K] bf16
  bf16_t* WkT = (bf16_t*)(ws + 4 * MK + 2 * KN);
  bf16_t* WvT = (bf16_t*)(ws + 4 * MK + 4 * KN);
  bf16_t* Qb  = (bf16_t*)(ws + 4 * MK + 6 * KN);             // [B,H,S,DH]
  bf16_t* Kb  = (bf16_t*)(ws + 6 * MK + 6 * KN);             // [B,H,S,DH]
  bf16_t* Vt  = (bf16_t*)(ws + 8 * MK + 6 * KN);             // [B,H,DH,S]

  // 1) convert activations to bf16 (vectorized x4)
  const int n4 = (int)(MK / 4);
  cvt_bf16_kernel<<<(n4 + 255) / 256, 256, 0, stream>>>(from_t, Xf, n4);
  cvt_bf16_kernel<<<(n4 + 255) / 256, 256, 0, stream>>>(to_t,   Xt, n4);

  // 2) transpose + convert weights (LDS-tiled, coalesced both sides)
  dim3 tgrid(DD / 32, DD / 32);
  wt_transpose_kernel<<<tgrid, 256, 0, stream>>>(Wq, WqT);
  wt_transpose_kernel<<<tgrid, 256, 0, stream>>>(Wk, WkT);
  wt_transpose_kernel<<<tgrid, 256, 0, stream>>>(Wv, WvT);

  // 3) QKV projections on the bf16 WMMA pipe
  dim3 ggrid(32, 8);           // M=4096/128, N=1024/128
  qkv_gemm_kernel<<<ggrid, 256, 0, stream>>>(Xf, WqT, bq, Qb, 0);
  qkv_gemm_kernel<<<ggrid, 256, 0, stream>>>(Xt, WkT, bk, Kb, 0);
  qkv_gemm_kernel<<<ggrid, 256, 0, stream>>>(Xt, WvT, bv, Vt, 1);

  // 4) flash attention
  dim3 agrid(SS / 128, BB * HH);
  flash_attn_kernel<<<agrid, 256, 0, stream>>>(Qb, Kb, Vt, amask, out);

  (void)in_sizes; (void)n_in; (void)out_size; (void)ws_size;
}
